// CUT_37082747634507
// MI455X (gfx1250) — matrix operand order repaired and checked by reading, not compiled, not run on hardware
//
#include <hip/hip_runtime.h>

// ---------------------------------------------------------------------------
// CDNA5 (gfx1250) wave32 WMMA implementation.
// Pipeline:  convert/pad logits + pre-convert/transpose all weights to f16
//            -> GEMM(w1,relu) -> GEMM(w2) -> GEMM(wq/wk/wv)
//            -> attention (WMMA scores + softmax + WMMA P@V) -> GEMM(wo)
//            (x2 branches) -> relevance GEMM (f32 out) -> per-batch top-64.
// GEMMs use double-buffered LDS tiles fed by async global->LDS when available.
// ---------------------------------------------------------------------------

typedef __attribute__((ext_vector_type(16))) _Float16    v16h;
typedef __attribute__((ext_vector_type(4)))  _Float16    v4h;
typedef __attribute__((ext_vector_type(8)))  float       v8f;
typedef __attribute__((ext_vector_type(4)))  float       v4f;
typedef __attribute__((ext_vector_type(4)))  unsigned int uint4v;
typedef int v4i_gcc __attribute__((vector_size(16)));    // builtin's b128 unit

#if __has_builtin(__builtin_amdgcn_global_load_async_to_lds_b128) && \
    __has_builtin(__builtin_amdgcn_s_wait_asynccnt)
#define USE_ASYNC_LDS 1
#else
#define USE_ASYNC_LDS 0
#endif

// copy 16B global -> LDS (per lane); async (ASYNCcnt) when the builtin exists
__device__ __forceinline__ void cp_async16(const _Float16* g, _Float16* l) {
#if USE_ASYNC_LDS
    __builtin_amdgcn_global_load_async_to_lds_b128(
        (__attribute__((address_space(1))) v4i_gcc*)(g),
        (__attribute__((address_space(3))) v4i_gcc*)(l), 0, 0);
#else
    *(uint4v*)l = *(const uint4v*)g;
#endif
}
__device__ __forceinline__ void cp_async_wait() {
#if USE_ASYNC_LDS
    __builtin_amdgcn_s_wait_asynccnt(0);
#endif
}

union FragU { v16h h; uint4v u[2]; };

// A-matrix fragment, 16x32 f16 (MxK).  LDS layout: [row][k] contiguous k.
__device__ __forceinline__ v16h ld_frag_a(const _Float16* p0, int ld) {
    int lane = threadIdx.x & 31;
    const _Float16* p = p0 + (lane & 15) * ld + ((lane >> 4) & 1) * 8;
    FragU f;
    f.u[0] = *(const uint4v*)(p);
    f.u[1] = *(const uint4v*)(p + 16);
    return f.h;
}

// B-matrix fragment, 32x16 f16 (KxN).  LDS layout: [col][k] contiguous k.
__device__ __forceinline__ v16h ld_frag_b(const _Float16* p0, int ld) {
    int lane = threadIdx.x & 31;
    const _Float16* p = p0 + (lane & 15) * ld + ((lane >> 4) & 1) * 8;
    FragU f;
    f.u[0] = *(const uint4v*)(p);
    f.u[1] = *(const uint4v*)(p + 16);
    return f.h;
}

__device__ __forceinline__ v8f wmma_f16(v16h a, v16h b, v8f c) {
    return __builtin_amdgcn_wmma_f32_16x16x32_f16(false, a, false, b, (short)0, c,
                                                  false, false);
}

// ---------------------------------------------------------------------------
// 0a) Convert obj_logits f32 [M,151] -> f16 [M,160] zero-padded.
// ---------------------------------------------------------------------------
__global__ __launch_bounds__(256) void k_convert_pad(const float* __restrict__ x,
                                                     _Float16* __restrict__ y,
                                                     int rows) {
    int i = blockIdx.x * 256 + threadIdx.x;
    int total = rows * 160;
    if (i >= total) return;
    int r = i / 160, c = i - r * 160;
    float v = (c < 151) ? x[(size_t)r * 151 + c] : 0.0f;
    y[i] = (_Float16)v;
}

// ---------------------------------------------------------------------------
// 0b) Convert weight W f32 [Kreal,D] -> transposed f16 WT [D][Kpad], zero-pad.
// ---------------------------------------------------------------------------
__global__ __launch_bounds__(256) void k_convert_wT(const float* __restrict__ W,
                                                    _Float16* __restrict__ WT,
                                                    int Kreal, int Kpad, int D) {
    int i = blockIdx.x * 256 + threadIdx.x;
    int total = D * Kpad;
    if (i >= total) return;
    int n = i / Kpad, k = i - n * Kpad;
    float v = (k < Kreal) ? W[(size_t)k * D + n] : 0.0f;
    WT[i] = (_Float16)v;
}

// ---------------------------------------------------------------------------
// 1) Tiled WMMA GEMM:  C[M,D]f16 = act(A[M,Kpad]f16 @ WT[D,Kpad]f16^T + bias)
//    256 threads (8 waves); tile 128(M) x 64(N); K chunk 32; 2x LDS buffers.
// ---------------------------------------------------------------------------
__global__ __launch_bounds__(256) void k_gemm(const _Float16* __restrict__ A,
                                              const _Float16* __restrict__ WT,
                                              const float* __restrict__ bias,
                                              _Float16* __restrict__ C,
                                              int lda, int Kpad, int D, int relu) {
    extern __shared__ char smem[];
    _Float16* As = (_Float16*)smem;              // 2 x [128][32]
    _Float16* Ws = As + 2 * 128 * 32;            // 2 x [64][32]
    const int bm = blockIdx.x * 128;
    const int bn = blockIdx.y * 64;
    const int t = threadIdx.x, wave = t >> 5, lane = t & 31;

    const int ar = t >> 1, ac = (t & 1) * 16;    // A-tile: 2 x 16B per thread
    const int wn = t >> 2, wc = (t & 3) * 8;     // W-tile: 1 x 16B per thread
    const _Float16* ag = A  + (size_t)(bm + ar) * lda  + ac;
    const _Float16* wg = WT + (size_t)(bn + wn) * Kpad + wc;

    auto stage = [&](int kc, int buf) {
        _Float16* as = As + buf * (128 * 32);
        _Float16* ws = Ws + buf * (64 * 32);
        cp_async16(ag + kc,     as + ar * 32 + ac);
        cp_async16(ag + kc + 8, as + ar * 32 + ac + 8);
        cp_async16(wg + kc,     ws + wn * 32 + wc);
    };

    v8f acc[4] = {};
    const int nIter = Kpad >> 5;
    stage(0, 0);
    cp_async_wait();
    __syncthreads();
    for (int i = 0; i < nIter; ++i) {
        const int buf = i & 1;
        if (i + 1 < nIter) stage((i + 1) << 5, buf ^ 1);   // overlap with WMMA
        const _Float16* as = As + buf * (128 * 32);
        const _Float16* ws = Ws + buf * (64 * 32);
        v16h af = ld_frag_a(as + wave * 16 * 32, 32);
#pragma unroll
        for (int j = 0; j < 4; ++j) {
            v16h bf = ld_frag_b(ws + j * 16 * 32, 32);
            acc[j] = wmma_f16(af, bf, acc[j]);
        }
        if (i + 1 < nIter) { cp_async_wait(); __syncthreads(); }
    }
    // epilogue: C/D layout -> row = r + 8*(lane>>4), col = lane&15 (per 16x16)
    int rbase = bm + wave * 16 + 8 * (lane >> 4);
#pragma unroll
    for (int j = 0; j < 4; ++j) {
        int col = bn + j * 16 + (lane & 15);
        float bv = bias ? bias[col] : 0.0f;
#pragma unroll
        for (int r = 0; r < 8; ++r) {
            float v = acc[j][r] + bv;
            if (relu) v = fmaxf(v, 0.0f);
            C[(size_t)(rbase + r) * D + col] = (_Float16)v;
        }
    }
}

// ---------------------------------------------------------------------------
// 2) Attention per (batch, head).  N=256 proposals, dh=32 (one K-chunk).
//    LDS: Qs/Ks 16KB each, Vt(32x256) 16KB, S strips 128KB f32, P strips 64KB.
// ---------------------------------------------------------------------------
__global__ __launch_bounds__(256) void k_attn(const _Float16* __restrict__ Q,
                                              const _Float16* __restrict__ K,
                                              const _Float16* __restrict__ V,
                                              _Float16* __restrict__ O) {
    extern __shared__ char smem[];
    _Float16* Qs = (_Float16*)smem;        // [256][32]
    _Float16* Ks = Qs + 256 * 32;          // [256][32]
    _Float16* Vt = Ks + 256 * 32;          // [32][256] = [n][k] transposed
    float*    S  = (float*)(Vt + 32 * 256);       // 8 waves x [16][256] f32
    _Float16* P  = (_Float16*)(S + 8 * 16 * 256); // 8 waves x [16][256] f16

    const int b = blockIdx.x, h = blockIdx.y;
    const int t = threadIdx.x, wave = t >> 5, lane = t & 31;
    const size_t base = (size_t)b * 256 * 256 + (size_t)h * 32;

    // load Q/K rows (async), V transposed scatter (thread t owns row t)
    {
        const _Float16* qr = Q + base + (size_t)t * 256;
        const _Float16* kr = K + base + (size_t)t * 256;
#pragma unroll
        for (int i = 0; i < 4; ++i) {
            cp_async16(qr + i * 8, Qs + t * 32 + i * 8);
            cp_async16(kr + i * 8, Ks + t * 32 + i * 8);
        }
        const _Float16* vr = V + base + (size_t)t * 256;
#pragma unroll
        for (int c = 0; c < 32; ++c) Vt[c * 256 + t] = vr[c];
        cp_async_wait();
    }
    __syncthreads();

    float* Sw = S + wave * 16 * 256;
    _Float16* Pw = P + wave * 16 * 256;
    const float scale = 0.17677669529663687f; // 1/sqrt(32)

    // scores: 16 WMMAs per wave, each K=32 == dh in one shot
    v16h qf = ld_frag_a(Qs + wave * 16 * 32, 32);
    for (int n0 = 0; n0 < 256; n0 += 16) {
        v16h kf = ld_frag_b(Ks + n0 * 32, 32);
        v8f acc = {};
        acc = wmma_f16(qf, kf, acc);
        int row = 8 * (lane >> 4), col = n0 + (lane & 15);
#pragma unroll
        for (int r = 0; r < 8; ++r) Sw[(row + r) * 256 + col] = acc[r] * scale;
    }
    __syncthreads();

    // softmax: 2 lanes per row (wave32 -> __shfl_xor 16), 128 cols each, 4-wide
    {
        int row = lane & 15, cb = (lane >> 4) * 128;
        v4f* sr4 = (v4f*)(Sw + row * 256 + cb);
        float m = -3.0e38f;
        for (int c = 0; c < 32; ++c) {
            v4f v = sr4[c];
            m = fmaxf(m, fmaxf(fmaxf(v[0], v[1]), fmaxf(v[2], v[3])));
        }
        m = fmaxf(m, __shfl_xor(m, 16));
        float sum = 0.0f;
        for (int c = 0; c < 32; ++c) {
            v4f v = sr4[c], e;
#pragma unroll
            for (int k = 0; k < 4; ++k) e[k] = __expf(v[k] - m);
            sum += (e[0] + e[1]) + (e[2] + e[3]);
            sr4[c] = e;
        }
        sum += __shfl_xor(sum, 16);
        float inv = 1.0f / sum;
        v4h* pr4 = (v4h*)(Pw + row * 256 + cb);
        for (int c = 0; c < 32; ++c) {
            v4f v = sr4[c];
            v4h hv;
#pragma unroll
            for (int k = 0; k < 4; ++k) hv[k] = (_Float16)(v[k] * inv);
            pr4[c] = hv;
        }
    }
    __syncthreads();

    // out = P(16x256) @ V(256x32): 8 K-chunks x 2 column tiles
    v8f oacc[2] = {};
    for (int kc = 0; kc < 256; kc += 32) {
        v16h pf = ld_frag_a(Pw + kc, 256);
#pragma unroll
        for (int j = 0; j < 2; ++j) {
            v16h vf = ld_frag_b(Vt + j * 16 * 256 + kc, 256);
            oacc[j] = wmma_f16(pf, vf, oacc[j]);
        }
    }
    int rbase = wave * 16 + 8 * (lane >> 4);
#pragma unroll
    for (int j = 0; j < 2; ++j) {
        int col = j * 16 + (lane & 15);
#pragma unroll
        for (int r = 0; r < 8; ++r)
            O[base + (size_t)(rbase + r) * 256 + col] = (_Float16)oacc[j][r];
    }
}

// ---------------------------------------------------------------------------
// 3) relevance[b] = subjE[b](256x256) @ objE[b]^T  -> f32 out.
//    objE rows [n][k] are already the contiguous-K layout B-frags need.
// ---------------------------------------------------------------------------
__global__ __launch_bounds__(256) void k_relevance(const _Float16* __restrict__ SE,
                                                   const _Float16* __restrict__ OE,
                                                   float* __restrict__ R) {
    extern __shared__ char smem[];
    _Float16* As = (_Float16*)smem;   // 2 x [128][32]
    _Float16* Bs = As + 2 * 128 * 32; // 2 x [64][32]
    const int b = blockIdx.x;
    const int bm = blockIdx.y * 128, bn = blockIdx.z * 64;
    const int t = threadIdx.x, wave = t >> 5, lane = t & 31;

    const int ar = t >> 1, ac = (t & 1) * 16;
    const int br_ = t >> 2, bc = (t & 3) * 8;
    const _Float16* ag = SE + (size_t)b * 65536 + (size_t)(bm + ar) * 256 + ac;
    const _Float16* bg = OE + (size_t)b * 65536 + (size_t)(bn + br_) * 256 + bc;

    auto stage = [&](int kc, int buf) {
        _Float16* as = As + buf * (128 * 32);
        _Float16* bs = Bs + buf * (64 * 32);
        cp_async16(ag + kc,     as + ar * 32 + ac);
        cp_async16(ag + kc + 8, as + ar * 32 + ac + 8);
        cp_async16(bg + kc,     bs + br_ * 32 + bc);
    };

    v8f acc[4] = {};
    stage(0, 0);
    cp_async_wait();
    __syncthreads();
    for (int i = 0; i < 8; ++i) {
        const int buf = i & 1;
        if (i + 1 < 8) stage((i + 1) << 5, buf ^ 1);
        const _Float16* as = As + buf * (128 * 32);
        const _Float16* bs = Bs + buf * (64 * 32);
        v16h af = ld_frag_a(as + wave * 16 * 32, 32);
#pragma unroll
        for (int j = 0; j < 4; ++j) {
            v16h bf = ld_frag_b(bs + j * 16 * 32, 32);
            acc[j] = wmma_f16(af, bf, acc[j]);
        }
        if (i + 1 < 8) { cp_async_wait(); __syncthreads(); }
    }
    float* Rb = R + (size_t)b * 65536;
    int rbase = bm + wave * 16 + 8 * (lane >> 4);
#pragma unroll
    for (int j = 0; j < 4; ++j) {
        int col = bn + j * 16 + (lane & 15);
#pragma unroll
        for (int r = 0; r < 8; ++r)
            Rb[(size_t)(rbase + r) * 256 + col] = acc[j][r];
    }
}

// ---------------------------------------------------------------------------
// 4) top-64 per batch: full 256KB row-block cached in LDS, iterative argmax
//    with 4-wide (b128) LDS scans.
// ---------------------------------------------------------------------------
__global__ __launch_bounds__(256) void k_topk64(const float* __restrict__ R,
                                                float* __restrict__ outIdx) {
    extern __shared__ char smem[];
    float* sv   = (float*)smem;        // 65536 f32
    float* redv = sv + 65536;          // 256
    int*   redi = (int*)(redv + 256);  // 256
    const int b = blockIdx.x, t = threadIdx.x;
    const float* Rb = R + (size_t)b * 65536;
    {
        const v4f* g4 = (const v4f*)Rb;
        v4f* s4 = (v4f*)sv;
        for (int i = t; i < 16384; i += 256) s4[i] = g4[i];
    }
    __syncthreads();
    float* ob = outIdx + (size_t)b * 128;
    for (int sel = 0; sel < 64; ++sel) {
        float bm = -3.4e38f; int bi = 0;
        const v4f* sv4 = (const v4f*)sv;
        for (int j = 0; j < 64; ++j) {
            int vi = t + j * 256;
            v4f v = sv4[vi];
#pragma unroll
            for (int e = 0; e < 4; ++e)
                if (v[e] > bm) { bm = v[e]; bi = vi * 4 + e; }
        }
        redv[t] = bm; redi[t] = bi;
        __syncthreads();
        for (int s = 128; s > 0; s >>= 1) {
            if (t < s) {
                bool take = (redv[t + s] > redv[t]) ||
                            (redv[t + s] == redv[t] && redi[t + s] < redi[t]);
                if (take) { redv[t] = redv[t + s]; redi[t] = redi[t + s]; }
            }
            __syncthreads();
        }
        if (t == 0) {
            int idx = redi[0];
            ob[sel * 2 + 0] = (float)(idx >> 8);
            ob[sel * 2 + 1] = (float)(idx & 255);
            sv[idx] = -3.4e38f;
        }
        __syncthreads();
    }
}

// ---------------------------------------------------------------------------
// Launch
// ---------------------------------------------------------------------------
extern "C" void kernel_launch(void* const* d_in, const int* in_sizes, int n_in,
                              void* d_out, int out_size, void* d_ws, size_t ws_size,
                              hipStream_t stream) {
    (void)in_sizes; (void)n_in; (void)out_size; (void)ws_size;
    const int B = 128, N = 256, D = 256, M = B * N;   // M = 32768
    const float* x = (const float*)d_in[0];            // obj_logits [B,N,151]
    // d_in[1] = num_pair_proposals (==64, hardcoded)
    float* out = (float*)d_out;                        // relevance then idx pairs

    // workspace carve-up (f16 staging, ~130 MB)
    char* w = (char*)d_ws;
    _Float16* Ain  = (_Float16*)w;  w += (size_t)M * 160 * 2;
    _Float16* H    = (_Float16*)w;  w += (size_t)M * D * 2;   // hidden / attn-out
    _Float16* E    = (_Float16*)w;  w += (size_t)M * D * 2;
    _Float16* Qb   = (_Float16*)w;  w += (size_t)M * D * 2;
    _Float16* Kb   = (_Float16*)w;  w += (size_t)M * D * 2;
    _Float16* Vb   = (_Float16*)w;  w += (size_t)M * D * 2;
    _Float16* embS = (_Float16*)w;  w += (size_t)M * D * 2;
    _Float16* embO = (_Float16*)w;  w += (size_t)M * D * 2;
    // pre-converted transposed f16 weights: per branch {w1T, w2T, wqT, wkT, wvT, woT}
    _Float16* WT[2][6];
    for (int br = 0; br < 2; ++br) {
        WT[br][0] = (_Float16*)w;  w += (size_t)D * 160 * 2;       // w1T [256][160]
        for (int m = 1; m < 6; ++m) { WT[br][m] = (_Float16*)w; w += (size_t)D * 256 * 2; }
    }

    const size_t GEMM_SMEM = 2 * (128 * 32 + 64 * 32) * sizeof(_Float16);  // 24 KB
    const size_t ATTN_SMEM = (3 * 256 * 32) * sizeof(_Float16)
                           + (8 * 16 * 256) * sizeof(float)
                           + (8 * 16 * 256) * sizeof(_Float16);            // 240 KB
    const size_t TOPK_SMEM = 65536 * 4 + 256 * 4 + 256 * 4;                // 258 KB

    // one shared padded-f16 copy of the logits
    k_convert_pad<<<(M * 160 + 255) / 256, 256, 0, stream>>>(x, Ain, M);

    // one-time weight conversion (transposed + zero-padded)
    for (int br = 0; br < 2; ++br) {
        const int base = 2 + br * 12;
        k_convert_wT<<<(D * 160 + 255) / 256, 256, 0, stream>>>(
            (const float*)d_in[base + 0], WT[br][0], 151, 160, D);
        for (int m = 1; m < 6; ++m)
            k_convert_wT<<<(D * 256 + 255) / 256, 256, 0, stream>>>(
                (const float*)d_in[base + 2 * m], WT[br][m], 256, 256, D);
    }

    dim3 ggrid(M / 128, D / 64);   // (256, 4)
    for (int br = 0; br < 2; ++br) {
        const int base = 2 + br * 12;
        const float* b1 = (const float*)d_in[base + 1];
        const float* b2 = (const float*)d_in[base + 3];
        const float* bq = (const float*)d_in[base + 5];
        const float* bk = (const float*)d_in[base + 7];
        const float* bv = (const float*)d_in[base + 9];
        const float* bo = (const float*)d_in[base + 11];
        _Float16* emb = br ? embO : embS;

        k_gemm<<<ggrid, 256, GEMM_SMEM, stream>>>(Ain, WT[br][0], b1, H, 160, 160, D, 1);
        k_gemm<<<ggrid, 256, GEMM_SMEM, stream>>>(H,   WT[br][1], b2, E, 256, 256, D, 0);
        k_gemm<<<ggrid, 256, GEMM_SMEM, stream>>>(E,   WT[br][2], bq, Qb, 256, 256, D, 0);
        k_gemm<<<ggrid, 256, GEMM_SMEM, stream>>>(E,   WT[br][3], bk, Kb, 256, 256, D, 0);
        k_gemm<<<ggrid, 256, GEMM_SMEM, stream>>>(E,   WT[br][4], bv, Vb, 256, 256, D, 0);
        k_attn<<<dim3(B, 8), 256, ATTN_SMEM, stream>>>(Qb, Kb, Vb, H);
        k_gemm<<<ggrid, 256, GEMM_SMEM, stream>>>(H,   WT[br][5], bo, emb, 256, 256, D, 0);
    }

    k_relevance<<<dim3(B, 2, 4), 256, GEMM_SMEM, stream>>>(embS, embO, out);
    k_topk64<<<B, 256, TOPK_SMEM, stream>>>(out, out + (size_t)B * 65536);
}
